// WSLFA_SA_5583457485355
// MI455X (gfx1250) — matrix operand order repaired and checked by reading, not compiled, not run on hardware
//
#include <hip/hip_runtime.h>
#include <hip/hip_bf16.h>
#include <stdint.h>

typedef __attribute__((ext_vector_type(16))) _Float16 v16h;
typedef __attribute__((ext_vector_type(8)))  _Float16 v8h;
typedef __attribute__((ext_vector_type(8)))  float    v8f;
typedef __attribute__((ext_vector_type(4)))  unsigned int u32x4;
typedef __attribute__((ext_vector_type(8)))  int      i32x8;
typedef __attribute__((ext_vector_type(4)))  int      i32x4;

union V16 { v16h v; v8h h[2]; };

constexpr int Bn    = 4;
constexpr int Pn    = 8192;
constexpr int Cn    = 64;
constexpr int Mn    = Pn / 4;      // 2048 centers per batch
constexpr int Kn    = 32;          // neighbors
constexpr int HID   = 128;
constexpr int OUTC  = 128;
constexpr int QC    = 64;
constexpr int CIN1  = Cn + 3;      // 67
constexpr int CINP1 = 96;          // padded to 3 x 32 for WMMA K-steps
constexpr long NSAMP = (long)Bn * Mn * Kn;   // 262144 rows
constexpr float EPSV = 1e-5f;

// ---------------------------------------------------------------------------
// Prep: pad/convert weights to f16 (K-contiguous rows), zero the BN stats.
// ---------------------------------------------------------------------------
__global__ void prep_kernel(const float* __restrict__ W1, const float* __restrict__ W2,
                            const float* __restrict__ Wq1,
                            _Float16* __restrict__ Wp1, _Float16* __restrict__ Wp2,
                            _Float16* __restrict__ WpQ1,
                            float* __restrict__ stats1, float* __restrict__ stats2) {
  int i = blockIdx.x * blockDim.x + threadIdx.x;
  if (i < HID * CINP1) {
    int o = i / CINP1, c = i % CINP1;
    Wp1[i] = (_Float16)(c < CIN1 ? W1[o * CIN1 + c] : 0.f);
  }
  if (i < OUTC * HID) Wp2[i]  = (_Float16)W2[i];
  if (i < QC * OUTC)  WpQ1[i] = (_Float16)Wq1[i];
  if (i < 2 * HID)  stats1[i] = 0.f;
  if (i < 2 * OUTC) stats2[i] = 0.f;
}

// ---------------------------------------------------------------------------
// KNN: one block per (b, m). All 8192 distances in LDS, then 32 stable
// argmin passes (lower-index tie-break == top_k order). Also emits centers.
// ---------------------------------------------------------------------------
__global__ __launch_bounds__(256) void knn_kernel(const float* __restrict__ xyz,
                                                  int* __restrict__ idxOut,
                                                  float* __restrict__ centersOut) {
  __shared__ float sdist[Pn];
  __shared__ float rv[256];
  __shared__ int   ri[256];
  int bm = blockIdx.x;
  int b = bm / Mn, m = bm % Mn;
  int tid = threadIdx.x;
  int cIdx = (int)(((long)m * (Pn - 1)) / (Mn - 1));   // floor(linspace)
  const float* xb = xyz + (size_t)b * Pn * 3;
  float cx = xb[cIdx * 3 + 0], cy = xb[cIdx * 3 + 1], cz = xb[cIdx * 3 + 2];
  if (tid < 3) centersOut[(size_t)bm * 3 + tid] = xb[cIdx * 3 + tid];
  float cc = cx * cx + cy * cy + cz * cz;
  for (int p = tid; p < Pn; p += 256) {
    float x = xb[p * 3 + 0], y = xb[p * 3 + 1], z = xb[p * 3 + 2];
    float xx  = x * x + y * y + z * z;
    float dot = cx * x + cy * y + cz * z;
    sdist[p] = cc + xx - 2.f * dot;                    // reference formula
  }
  __syncthreads();
  for (int t = 0; t < Kn; ++t) {
    float best = 3.4e38f; int bi = Pn;
    for (int p = tid; p < Pn; p += 256) {
      float d = sdist[p];
      if (d < best || (d == best && p < bi)) { best = d; bi = p; }
    }
    rv[tid] = best; ri[tid] = bi;
    __syncthreads();
    for (int s = 128; s > 0; s >>= 1) {
      if (tid < s) {
        float ov = rv[tid + s]; int oi = ri[tid + s];
        if (ov < rv[tid] || (ov == rv[tid] && oi < ri[tid])) { rv[tid] = ov; ri[tid] = oi; }
      }
      __syncthreads();
    }
    if (tid == 0) {
      idxOut[(size_t)bm * Kn + t] = ri[0];
      sdist[ri[0]] = 3.4e38f;
    }
    __syncthreads();
  }
}

// ---------------------------------------------------------------------------
// Gather: build A matrix rows [local_xyz(3) | feats(64) | pad(29)] as f16.
// ---------------------------------------------------------------------------
__global__ void gather_kernel(const float* __restrict__ xyz, const float* __restrict__ feats,
                              const float* __restrict__ centers, const int* __restrict__ idx,
                              _Float16* __restrict__ Acat) {
  long gid = (long)blockIdx.x * blockDim.x + threadIdx.x;
  long total = NSAMP * CINP1;
  if (gid >= total) return;
  long n = gid / CINP1;
  int  c = (int)(gid % CINP1);
  int  b = (int)(n >> 16);              // n / (Mn*Kn), Mn*Kn == 65536
  long mk = n & (long)(Mn * Kn - 1);
  int  m = (int)(mk >> 5);
  int  p = idx[n];
  float v;
  if (c < 3)          v = xyz[((size_t)b * Pn + p) * 3 + c] - centers[((size_t)b * Mn + m) * 3 + c];
  else if (c < CIN1)  v = feats[(size_t)b * Cn * Pn + (size_t)(c - 3) * Pn + p];
  else                v = 0.f;
  Acat[gid] = (_Float16)v;
}

// ---------------------------------------------------------------------------
// Stage a weight block (COUT x CINP f16, row-major) into LDS.
// Preferred path: one Tensor Data Mover descriptor issued by wave 0
// (1-D tile, data_size = 4B), completion via s_wait_tensorcnt.
// ---------------------------------------------------------------------------
template <int CINP, int COUT, int NTHR>
__device__ __forceinline__ void stage_weights(const _Float16* __restrict__ Wg,
                                              _Float16* sW, int tid, int wave) {
#if defined(__has_builtin) && __has_builtin(__builtin_amdgcn_tensor_load_to_lds)
  if (wave == 0) {
    constexpr unsigned NDW = (unsigned)(COUT * CINP * 2 / 4);   // dwords, fits 16 bits
    unsigned long long ga = (unsigned long long)(uintptr_t)Wg;
    unsigned int lds = (unsigned int)(uintptr_t)sW;             // LDS offset = addr[31:0]
    u32x4 g0;
    g0[0] = 1u;                                        // count=1, user descriptor
    g0[1] = lds;                                       // lds_addr
    g0[2] = (unsigned)(ga & 0xFFFFFFFFu);              // global_addr[31:0]
    g0[3] = (unsigned)((ga >> 32) & 0x1FFFFFFu) | (2u << 30);   // addr[56:32] | type=2
    i32x8 g1;
    g1[0] = (2 << 16);                                 // data_size = 4B
    g1[1] = (int)((NDW & 0xFFFFu) << 16);              // tensor_dim0[15:0] @ bits 63:48
    g1[2] = (int)((NDW >> 16) & 0xFFFFu) | (1 << 16);  // tensor_dim0 hi | tensor_dim1 = 1
    g1[3] = (int)(NDW << 16);                          // tile_dim0 @ bits 127:112
    g1[4] = 1;                                         // tile_dim1 = 1
    g1[5] = (int)NDW;                                  // tensor_dim0_stride
    g1[6] = 0;
    g1[7] = 0;
    i32x4 z4 = {0, 0, 0, 0};
#if __clang_major__ >= 23
    i32x8 z8 = {0, 0, 0, 0, 0, 0, 0, 0};
    __builtin_amdgcn_tensor_load_to_lds(g0, g1, z4, z4, z8, 0);
#else
    __builtin_amdgcn_tensor_load_to_lds(g0, g1, z4, z4, 0);
#endif
    __builtin_amdgcn_s_wait_tensorcnt(0);
  }
#else
  for (int i = tid; i < COUT * CINP; i += NTHR) sW[i] = Wg[i];
#endif
}

// ---------------------------------------------------------------------------
// WMMA GEMM: Y[n,o] = sum_c A[n,c] * W[o,c] (+bias). W staged in LDS via TDM.
// Each wave owns one 16-wide output-channel tile, block walks NITER n-tiles.
// DO_STATS: per-channel sum/sumsq accumulated in registers across tiles,
//           pair-reduced with shfl_xor(16), 2 LDS atomics/lane, then global.
// !DO_STATS: fused ReLU (q1 branch).
// ---------------------------------------------------------------------------
template <int CINP, int COUT, bool DO_STATS, int NITER>
__global__ void gemm_kernel(const _Float16* __restrict__ A, const _Float16* __restrict__ Wg,
                            const float* __restrict__ bias, _Float16* __restrict__ Y,
                            float* __restrict__ stats) {
  constexpr int WAVES = COUT / 16;
  constexpr int NTHR  = WAVES * 32;
  constexpr int KSTEPS = CINP / 32;
  __shared__ __align__(16) _Float16 sW[COUT * CINP];
  __shared__ float sred[2 * COUT];
  int tid = threadIdx.x;
  int lane = tid & 31, wave = tid >> 5;

  stage_weights<CINP, COUT, NTHR>(Wg, sW, tid, wave);
  if (DO_STATS) for (int i = tid; i < 2 * COUT; i += NTHR) sred[i] = 0.f;
  __syncthreads();

  int half_ = lane >> 4, r = lane & 15;
  int ot = wave * 16;

  // B fragments (32x16 K-major): lane col = ot+r, half selects K {0-7,16-23} vs {8-15,24-31}
  V16 bfrag[KSTEPS];
  for (int kk = 0; kk < KSTEPS; ++kk) {
    const _Float16* wp = &sW[(size_t)(ot + r) * CINP + kk * 32 + (half_ << 3)];
    bfrag[kk].h[0] = *(const v8h*)wp;
    bfrag[kk].h[1] = *(const v8h*)(wp + 16);
  }

  int n = ot + r;                 // output channel for this lane
  float bv = bias[n];
  float ssum = 0.f, ssq = 0.f;    // register-resident BN stats

  for (int t = 0; t < NITER; ++t) {
    long tileM = ((long)blockIdx.x * NITER + t) * 16;
    const _Float16* arow = A + (size_t)(tileM + r) * CINP;
    __builtin_prefetch(arow + (size_t)16 * CINP, 0, 0);   // next tile -> global_prefetch_b8
    v8f acc = {0.f, 0.f, 0.f, 0.f, 0.f, 0.f, 0.f, 0.f};
    for (int kk = 0; kk < KSTEPS; ++kk) {
      // A fragment (16x32): half selects K {0-7,16-23} vs {8-15,24-31}
      V16 a;
      const _Float16* ap = arow + kk * 32 + (half_ << 3);
      a.h[0] = *(const v8h*)ap;
      a.h[1] = *(const v8h*)(ap + 16);
      acc = __builtin_amdgcn_wmma_f32_16x16x32_f16(false, a.v, false, bfrag[kk].v,
                                                   (short)0, acc, false, false);
    }
    long mbase = tileM + (half_ << 3);
    for (int v = 0; v < 8; ++v) {
      float y = acc[v] + bv;
      if (DO_STATS) {
        ssum += y;
        ssq  += y * y;
      } else {
        y = fmaxf(y, 0.f);                     // fused ReLU (q1)
      }
      Y[(size_t)(mbase + v) * COUT + n] = (_Float16)y;
    }
  }
  if (DO_STATS) {
    // lanes L and L+16 hold the same channel: pair-reduce, then one atomic each
    ssum += __shfl_xor(ssum, 16, 32);
    ssq  += __shfl_xor(ssq, 16, 32);
    if (half_ == 0) {
      atomicAdd(&sred[n], ssum);               // ds_add_f32
      atomicAdd(&sred[COUT + n], ssq);
    }
    __syncthreads();
    for (int i = tid; i < 2 * COUT; i += NTHR) atomicAdd(&stats[i], sred[i]);
  }
}

// ---------------------------------------------------------------------------
// BatchNorm (train-mode, global batch stats) + ReLU, in place on f16 buffer.
// ---------------------------------------------------------------------------
__global__ void bn_relu_kernel(_Float16* __restrict__ Y, const float* __restrict__ stats,
                               const float* __restrict__ g, const float* __restrict__ be,
                               long nelems, int cout, float inv_ntot) {
  long i = (long)blockIdx.x * blockDim.x + threadIdx.x;
  if (i >= nelems) return;
  int o = (int)(i & (long)(cout - 1));     // cout is a power of two
  float mean = stats[o] * inv_ntot;
  float var  = stats[cout + o] * inv_ntot - mean * mean;
  float sc   = g[o] * rsqrtf(var + EPSV);
  float x = (float)Y[i];
  float y = (x - mean) * sc + be[o];
  Y[i] = (_Float16)fmaxf(y, 0.f);
}

// ---------------------------------------------------------------------------
// Fused q2-dot + softmax over K (one wave32 = one (b,m)) + weighted pooling.
// ---------------------------------------------------------------------------
__global__ __launch_bounds__(128) void attn_agg_kernel(const _Float16* __restrict__ Q1,
                                                       const _Float16* __restrict__ FL,
                                                       const float* __restrict__ Wq2,
                                                       const float* __restrict__ bq2,
                                                       float* __restrict__ outAgg) {
  __shared__ float sw[Kn];
  int bm = blockIdx.x;
  int tid = threadIdx.x;
  long base = (long)bm * Kn;
  if (tid < Kn) {
    const _Float16* q = Q1 + (base + tid) * QC;
    float acc = bq2[0];
    for (int o = 0; o < QC; ++o) acc += (float)q[o] * Wq2[o];
    float mx = acc;
    for (int s = 16; s > 0; s >>= 1) mx = fmaxf(mx, __shfl_xor(mx, s, 32));
    float e = __expf(acc - mx);
    float sm = e;
    for (int s = 16; s > 0; s >>= 1) sm += __shfl_xor(sm, s, 32);
    sw[tid] = e / sm;
  }
  __syncthreads();
  int o = tid;                               // 128 channels
  float acc = 0.f;
  for (int k = 0; k < Kn; ++k)
    acc += (float)FL[(base + k) * OUTC + o] * sw[k];
  int b = bm / Mn, m = bm % Mn;
  outAgg[((size_t)b * OUTC + o) * Mn + m] = acc;
}

// ---------------------------------------------------------------------------
extern "C" void kernel_launch(void* const* d_in, const int* in_sizes, int n_in,
                              void* d_out, int out_size, void* d_ws, size_t ws_size,
                              hipStream_t stream) {
  (void)in_sizes; (void)n_in; (void)out_size; (void)ws_size;
  const float* xyz   = (const float*)d_in[0];
  const float* feats = (const float*)d_in[1];
  const float* W1  = (const float*)d_in[2];
  const float* b1  = (const float*)d_in[3];
  const float* g1  = (const float*)d_in[4];
  const float* be1 = (const float*)d_in[5];
  const float* W2  = (const float*)d_in[6];
  const float* b2  = (const float*)d_in[7];
  const float* g2  = (const float*)d_in[8];
  const float* be2 = (const float*)d_in[9];
  const float* Wq1 = (const float*)d_in[10];
  const float* bq1 = (const float*)d_in[11];
  const float* Wq2 = (const float*)d_in[12];
  const float* bq2 = (const float*)d_in[13];

  float* out = (float*)d_out;
  float* centersOut = out;                          // (B, M, 3)
  float* aggOut = out + (size_t)Bn * Mn * 3;        // (B, OUTC, M)

  char* ws = (char*)d_ws;
  size_t off = 0;
  auto take = [&](size_t bytes) { void* p = ws + off; off = (off + bytes + 255) & ~(size_t)255; return p; };
  int*      idxBuf = (int*)take((size_t)NSAMP * sizeof(int));
  _Float16* Acat   = (_Float16*)take((size_t)NSAMP * CINP1 * 2);
  _Float16* Y1     = (_Float16*)take((size_t)NSAMP * HID * 2);
  _Float16* Y2     = (_Float16*)take((size_t)NSAMP * OUTC * 2);
  _Float16* Q1     = (_Float16*)take((size_t)NSAMP * QC * 2);
  _Float16* Wp1    = (_Float16*)take((size_t)HID * CINP1 * 2);
  _Float16* Wp2    = (_Float16*)take((size_t)OUTC * HID * 2);
  _Float16* WpQ1   = (_Float16*)take((size_t)QC * OUTC * 2);
  float*    stats1 = (float*)take(2 * HID * sizeof(float));
  float*    stats2 = (float*)take(2 * OUTC * sizeof(float));

  prep_kernel<<<64, 256, 0, stream>>>(W1, W2, Wq1, Wp1, Wp2, WpQ1, stats1, stats2);
  knn_kernel<<<Bn * Mn, 256, 0, stream>>>(xyz, idxBuf, centersOut);
  {
    long total = NSAMP * CINP1;
    gather_kernel<<<(int)((total + 255) / 256), 256, 0, stream>>>(xyz, feats, centersOut, idxBuf, Acat);
  }
  constexpr int NITER = 8;
  int gemm_blocks = (int)(NSAMP / 16 / NITER);      // 2048
  gemm_kernel<CINP1, HID, true, NITER><<<gemm_blocks, (HID / 16) * 32, 0, stream>>>(Acat, Wp1, b1, Y1, stats1);
  {
    long nel = NSAMP * HID;
    bn_relu_kernel<<<(int)((nel + 255) / 256), 256, 0, stream>>>(Y1, stats1, g1, be1, nel, HID, 1.f / (float)NSAMP);
  }
  gemm_kernel<HID, OUTC, true, NITER><<<gemm_blocks, (OUTC / 16) * 32, 0, stream>>>(Y1, Wp2, b2, Y2, stats2);
  {
    long nel = NSAMP * OUTC;
    bn_relu_kernel<<<(int)((nel + 255) / 256), 256, 0, stream>>>(Y2, stats2, g2, be2, nel, OUTC, 1.f / (float)NSAMP);
  }
  gemm_kernel<OUTC, QC, false, NITER><<<gemm_blocks, (QC / 16) * 32, 0, stream>>>(Y2, WpQ1, bq1, Q1, nullptr);
  attn_agg_kernel<<<Bn * Mn, 128, 0, stream>>>(Q1, Y2, Wq2, bq2, aggOut);
}
